// ESModel_22522808500727
// MI455X (gfx1250) — matrix operand order repaired and checked by reading, not compiled, not run on hardware
//
#include <hip/hip_runtime.h>

// ---------------- problem constants ----------------
#define D_IN   1024
#define HID    2048
#define D_OUT  10
#define POP    40
#define NROW   256
#define STD    0.1f
#define S1 (D_IN * HID)          // 2097152  (W1 noise)
#define S2 (S1 + HID)            // 2099200  (after b1 noise)
#define S3 (S2 + HID * D_OUT)    // 2119680  (after W2 noise)
#define NP (S3 + D_OUT)          // 2119690  total params (even; p*NP*4 is 0 mod 8, 8 mod 16 for odd p)

typedef __bf16 bf16;
typedef __attribute__((ext_vector_type(16))) __bf16 v16bf;
typedef __attribute__((ext_vector_type(8)))  __bf16 v8bf;
typedef __attribute__((ext_vector_type(8)))  float  v8f;
typedef __attribute__((ext_vector_type(4)))  float  v4f;
typedef __attribute__((ext_vector_type(2)))  float  v2f;
typedef __attribute__((ext_vector_type(4)))  unsigned v4u;
typedef __attribute__((ext_vector_type(2)))  unsigned v2u;

// pack two f32 into one dword of two bf16 (round-to-nearest-even).
// Single-instruction path via v_cvt_pk_bf16_f32 (gfx1250 VALU bf16 converts).
// Fallback (if ever needed): the manual RNE bit-twiddle shown in the comment.
static __device__ __forceinline__ unsigned pack2bf(float a, float b) {
    unsigned r;
    asm("v_cvt_pk_bf16_f32 %0, %1, %2" : "=v"(r) : "v"(a), "v"(b));
    return r;
    // unsigned ua = __builtin_bit_cast(unsigned, a), ub = __builtin_bit_cast(unsigned, b);
    // return ((ua + 0x7FFFu + ((ua >> 16) & 1u)) >> 16)
    //      | ((ub + 0x7FFFu + ((ub >> 16) & 1u)) & 0xFFFF0000u);
}

// full stage (prologue): A tile 128x32, B tile 32x128, f32 -> bf16 into LDS
static __device__ __forceinline__ void stage_full(
    const float* __restrict__ A, const float* __restrict__ B,
    int row0, int n0, int k0, bf16* ldsA, bf16* ldsB, int t)
{
    #pragma unroll
    for (int idx = t; idx < 512; idx += 256) {
        int r = idx >> 2, c8 = (idx & 3) * 8;
        const float* src = &A[(size_t)(row0 + r) * D_IN + k0 + c8];
        v4f f0 = *(const v4f*)src;
        v4f f1 = *(const v4f*)(src + 4);
        v4u pk = { pack2bf(f0.x, f0.y), pack2bf(f0.z, f0.w),
                   pack2bf(f1.x, f1.y), pack2bf(f1.z, f1.w) };
        *(v4u*)&ldsA[r * 32 + c8] = pk;
    }
    #pragma unroll
    for (int idx = t; idx < 1024; idx += 256) {
        int kk = idx >> 5, c4 = (idx & 31) * 4;
        const float* src = &B[(size_t)(k0 + kk) * HID + n0 + c4];
        v2f f0 = *(const v2f*)src;
        v2f f1 = *(const v2f*)(src + 2);
        v2u pk = { pack2bf(f0.x, f0.y), pack2bf(f1.x, f1.y) };
        *(v2u*)&ldsB[kk * 128 + c4] = pk;
    }
}

// ---------------------------------------------------------------------------
// Double-buffered GEMM tile: C(128x128 f32) += A(128xK bf16) * B(Kx128 bf16).
// 8 waves (4 row x 2 col), each wave 2x4 v_wmma_f32_16x16x32_bf16 accumulators.
// Pipeline per K-step (one barrier each):
//   global loads (tile i+1) -> regs;  frag ds_loads + WMMA (tile i);
//   pack + ds_store (tile i+1).
// The ds_stores are issued AFTER the frag ds_loads on purpose: DS ops complete
// in order per wave, so this keeps the WMMA's s_wait_dscnt off the next tile's
// global-load latency.
// ---------------------------------------------------------------------------
static __device__ __forceinline__ void gemm_tile_bf16_db(
    const float* __restrict__ A,   // [NROW][D_IN] row-major, 16B-aligned rows
    const float* __restrict__ B,   // [D_IN][HID]  row-major, 8B-aligned
    int row0, int n0,
    bf16* lds0A, bf16* lds0B, bf16* lds1A, bf16* lds1B,
    v8f acc[2][4])
{
    const int t    = threadIdx.x;
    const int lane = t & 31;
    const int w    = t >> 5;
    const int wr   = w >> 1;       // 0..3 -> rows 32*wr
    const int wc   = w & 1;        // 0..1 -> cols 64*wc
    const int m    = lane & 15;    // A row within 16-row tile
    const int kh   = lane >> 4;    // A K-half select

    v8f zero = {};
    #pragma unroll
    for (int i = 0; i < 2; ++i)
        #pragma unroll
        for (int j = 0; j < 4; ++j) acc[i][j] = zero;

    stage_full(A, B, row0, n0, 0, lds0A, lds0B, t);

    for (int k0 = 0; k0 < D_IN; k0 += 32) {
        const int cur = (k0 >> 5) & 1;
        bf16* curA = cur ? lds1A : lds0A;
        bf16* curB = cur ? lds1B : lds0B;
        bf16* nxtA = cur ? lds0A : lds1A;
        bf16* nxtB = cur ? lds0B : lds1B;
        const bool more = (k0 + 32) < D_IN;
        __syncthreads();

        // (a) global loads for the next tile into registers
        v4f pa[2][2]; v2f pb[4][2];
        if (more) {
            #pragma unroll
            for (int u = 0; u < 2; ++u) {
                int idx = t + 256 * u;
                int r = idx >> 2, c8 = (idx & 3) * 8;
                const float* src = &A[(size_t)(row0 + r) * D_IN + (k0 + 32) + c8];
                pa[u][0] = *(const v4f*)src;
                pa[u][1] = *(const v4f*)(src + 4);
            }
            #pragma unroll
            for (int u = 0; u < 4; ++u) {
                int idx = t + 256 * u;
                int kk = idx >> 5, c4 = (idx & 31) * 4;
                const float* src = &B[(size_t)(k0 + 32 + kk) * HID + n0 + c4];
                pb[u][0] = *(const v2f*)src;
                pb[u][1] = *(const v2f*)(src + 2);
            }
        }

        // (b) fragments + WMMA on the current tile
        // A frag: lane<16 holds K {0..7,16..23}, lane>=16 holds K {8..15,24..31}
        v16bf af[2];
        #pragma unroll
        for (int i = 0; i < 2; ++i) {
            int rowl = 32 * wr + 16 * i + m;
            v8bf lo = *(const v8bf*)&curA[rowl * 32 + kh * 8];
            v8bf hi = *(const v8bf*)&curA[rowl * 32 + 16 + kh * 8];
            #pragma unroll
            for (int e = 0; e < 8; ++e) { af[i][e] = lo[e]; af[i][8 + e] = hi[e]; }
        }
        // B frag: lane%16 = N, lane<16 holds K {0..7,16..23}, lane>=16 holds K {8..15,24..31}
        v16bf bfv[4];
        #pragma unroll
        for (int j = 0; j < 4; ++j) {
            int coll = 64 * wc + 16 * j + m;
            #pragma unroll
            for (int e = 0; e < 8; ++e) {
                bfv[j][e] = curB[(kh * 8 + e) * 128 + coll];
                bfv[j][8 + e] = curB[(16 + kh * 8 + e) * 128 + coll];
            }
        }
        #pragma unroll
        for (int i = 0; i < 2; ++i)
            #pragma unroll
            for (int j = 0; j < 4; ++j)
                acc[i][j] = __builtin_amdgcn_wmma_f32_16x16x32_bf16(
                    false, af[i], false, bfv[j], (short)0, acc[i][j], false, false);

        // (c) pack + ds_store of the next tile
        if (more) {
            #pragma unroll
            for (int u = 0; u < 2; ++u) {
                int idx = t + 256 * u;
                int r = idx >> 2, c8 = (idx & 3) * 8;
                v4u pk = { pack2bf(pa[u][0].x, pa[u][0].y), pack2bf(pa[u][0].z, pa[u][0].w),
                           pack2bf(pa[u][1].x, pa[u][1].y), pack2bf(pa[u][1].z, pa[u][1].w) };
                *(v4u*)&nxtA[r * 32 + c8] = pk;
            }
            #pragma unroll
            for (int u = 0; u < 4; ++u) {
                int idx = t + 256 * u;
                int kk = idx >> 5, c4 = (idx & 31) * 4;
                v2u pk = { pack2bf(pb[u][0].x, pb[u][0].y), pack2bf(pb[u][1].x, pb[u][1].y) };
                *(v2u*)&nxtB[kk * 128 + c4] = pk;
            }
        }
    }
}

// ---------------------------------------------------------------------------
// Kernel 1: Hb = x @ W1  (f32 out)
// ---------------------------------------------------------------------------
__global__ void __launch_bounds__(256)
k_base_hidden(const float* __restrict__ x, const float* __restrict__ W1,
              float* __restrict__ Hb)
{
    __shared__ bf16 lds0A[128 * 32];
    __shared__ bf16 lds0B[32 * 128];
    __shared__ bf16 lds1A[128 * 32];
    __shared__ bf16 lds1B[32 * 128];
    const int n0   = blockIdx.x * 128;
    const int row0 = blockIdx.y * 128;
    v8f acc[2][4];
    gemm_tile_bf16_db(x, W1, row0, n0, lds0A, lds0B, lds1A, lds1B, acc);

    const int lane = threadIdx.x & 31;
    const int w = threadIdx.x >> 5, wr = w >> 1, wc = w & 1;
    #pragma unroll
    for (int i = 0; i < 2; ++i)
        #pragma unroll
        for (int j = 0; j < 4; ++j) {
            int coll  = 64 * wc + 16 * j + (lane & 15);
            int rbase = 32 * wr + 16 * i + 8 * (lane >> 4);
            #pragma unroll
            for (int r = 0; r < 8; ++r)
                Hb[(size_t)(row0 + rbase + r) * HID + (n0 + coll)] = acc[i][j][r];
        }
}

// ---------------------------------------------------------------------------
// Kernel 2: per (p, 128x128 tile): Hn = x @ N1_p; then fused layer-2 partials:
//   predpart[((p*2+s)*16 + colblk)][row][o] =
//       sum over this tile's 128 hidden cols of relu(h±) * (W2 ± σN2)
// W2/N2 tiles are contiguous byte-copies -> CDNA5 async global->LDS DMA
// (ASYNCcnt), issued before the GEMM so the copy hides under the K-loop.
// hbuf aliases the (dead) GEMM stage buffers. Deterministic (no atomics).
// ---------------------------------------------------------------------------
__global__ void __launch_bounds__(256)
k_noise_fused(const float* __restrict__ x, const float* __restrict__ Hb,
              const float* __restrict__ b1, const float* __restrict__ W2,
              const float* __restrict__ noise, float* __restrict__ predpart)
{
    __shared__ __align__(16) char smem[48128];
    bf16*  lds0A  = (bf16*)smem;                  //  8192 B
    bf16*  lds0B  = (bf16*)(smem + 8192);         //  8192 B
    bf16*  lds1A  = (bf16*)(smem + 16384);        //  8192 B
    bf16*  lds1B  = (bf16*)(smem + 24576);        //  8192 B
    float* hbuf   = (float*)smem;                 // 32768 B, aliases stage bufs
    float* w2t    = (float*)(smem + 32768);       //  5120 B (128 x 10)
    float* n2t    = (float*)(smem + 37888);       //  5120 B (128 x 10)
    float* predsh = (float*)(smem + 43008);       //  5120 B

    const int n0   = blockIdx.x * 128;
    const int row0 = blockIdx.y * 128;
    const int p    = blockIdx.z;
    const size_t pbase = (size_t)p * NP;
    const float* N1  = noise + pbase;
    const float* nb1 = noise + pbase + S1;
    const float* N2  = noise + pbase + S2;

    const int t = threadIdx.x;

    // Async-DMA the layer-2 weight tiles (contiguous 5120 B each) into LDS.
    // 640 x b64 chunks per tile; 8B alignment holds for any p (p*NP*4 ≡ 0 mod 8).
    {
        unsigned w2off = (unsigned)(uintptr_t)w2t;
        unsigned n2off = (unsigned)(uintptr_t)n2t;
        unsigned long long w2src = (unsigned long long)(uintptr_t)(W2 + (size_t)n0 * D_OUT);
        unsigned long long n2src = (unsigned long long)(uintptr_t)(N2 + (size_t)n0 * D_OUT);
        for (int idx = t; idx < 640; idx += 256) {
            unsigned long long ga = w2src + (unsigned long long)idx * 8ull;
            unsigned la = w2off + (unsigned)idx * 8u;
            asm volatile("global_load_async_to_lds_b64 %0, %1, off"
                         :: "v"(la), "v"(ga) : "memory");
            unsigned long long gb = n2src + (unsigned long long)idx * 8ull;
            unsigned lb = n2off + (unsigned)idx * 8u;
            asm volatile("global_load_async_to_lds_b64 %0, %1, off"
                         :: "v"(lb), "v"(gb) : "memory");
        }
    }

    v8f acc[2][4];
    gemm_tile_bf16_db(x, N1, row0, n0, lds0A, lds0B, lds1A, lds1B, acc);

    // fence async DMA; barrier also protects hbuf's aliasing of stage buffers
    asm volatile("s_wait_asynccnt 0x0" ::: "memory");
    __syncthreads();

    const int lane = t & 31;
    const int w = t >> 5, wr = w >> 1, wc = w & 1;
    const int m = lane & 15, kh = lane >> 4;
    const int rl   = t >> 1;   // reduction row (0..127)
    const int half = t & 1;    // which 32-col chunk of the 64-col hbuf

    // hoist Hb / b1 / b1-noise for this wave's columns (shared by both signs)
    float hbv[2][4][8], b1v[4], nb1v[4];
    #pragma unroll
    for (int j = 0; j < 4; ++j) {
        int colg = n0 + 64 * wc + 16 * j + m;
        b1v[j]  = b1[colg];
        nb1v[j] = nb1[colg];
        #pragma unroll
        for (int i = 0; i < 2; ++i)
            #pragma unroll
            for (int r = 0; r < 8; ++r)
                hbv[i][j][r] = Hb[(size_t)(row0 + 32 * wr + 16 * i + 8 * kh + r) * HID + colg];
    }

    for (int s = 0; s < 2; ++s) {
        const float par = s ? -STD : STD;   // signed sigma
        float partial[D_OUT];
        #pragma unroll
        for (int o = 0; o < D_OUT; ++o) partial[o] = 0.0f;

        for (int ch2 = 0; ch2 < 2; ++ch2) {   // which 64-col half of the tile
            if (wc == ch2) {                   // wave-uniform branch
                #pragma unroll
                for (int i = 0; i < 2; ++i)
                    #pragma unroll
                    for (int j = 0; j < 4; ++j) {
                        float bv = b1v[j] + par * nb1v[j];
                        int c64 = 16 * j + m;
                        #pragma unroll
                        for (int r = 0; r < 8; ++r) {
                            int rr = 32 * wr + 16 * i + 8 * kh + r;
                            float h = hbv[i][j][r] + par * acc[i][j][r] + bv;
                            hbuf[rr * 64 + c64] = fmaxf(h, 0.0f);
                        }
                    }
            }
            __syncthreads();
            for (int c = 32 * half; c < 32 * half + 32; ++c) {
                float hv = hbuf[rl * 64 + c];
                int cg = 64 * ch2 + c;
                #pragma unroll
                for (int o = 0; o < D_OUT; ++o)
                    partial[o] += hv * (w2t[cg * D_OUT + o] + par * n2t[cg * D_OUT + o]);
            }
            __syncthreads();
        }
        // combine the two threads covering each row (fixed order -> deterministic)
        if (half == 1) {
            #pragma unroll
            for (int o = 0; o < D_OUT; ++o) predsh[rl * D_OUT + o] = partial[o];
        }
        __syncthreads();
        if (half == 0) {
            float* pd = predpart +
                ((((size_t)(p * 2 + s) * 16 + blockIdx.x) * NROW) + (row0 + rl)) * D_OUT;
            #pragma unroll
            for (int o = 0; o < D_OUT; ++o)
                pd[o] = partial[o] + predsh[rl * D_OUT + o];
        }
        __syncthreads();
    }
}

// ---------------------------------------------------------------------------
// Kernel 3: losses[p*2+s] = mean((pred + b2 ± σnb2 - y)^2), fixed reduce order
// ---------------------------------------------------------------------------
__global__ void __launch_bounds__(256)
k_losses(const float* __restrict__ predpart, const float* __restrict__ y,
         const float* __restrict__ b2, const float* __restrict__ noise,
         float* __restrict__ losses)
{
    const int ps = blockIdx.x;          // 0..79
    const int p  = ps >> 1, s = ps & 1;
    const float par = s ? -STD : STD;
    const int r = threadIdx.x;          // row 0..255
    const float* nb2 = noise + (size_t)p * NP + S3;

    float accum = 0.0f;
    #pragma unroll
    for (int o = 0; o < D_OUT; ++o) {
        float pv = 0.0f;
        for (int cb = 0; cb < 16; ++cb)
            pv += predpart[(((size_t)ps * 16 + cb) * NROW + r) * D_OUT + o];
        float d = pv + b2[o] + par * nb2[o] - y[(size_t)r * D_OUT + o];
        accum += d * d;
    }
    __shared__ float red[256];
    red[r] = accum;
    __syncthreads();
    for (int off = 128; off > 0; off >>= 1) {
        if (r < off) red[r] += red[r + off];
        __syncthreads();
    }
    if (r == 0) losses[ps] = red[0] * (1.0f / (float)(NROW * D_OUT));
}

// ---------------------------------------------------------------------------
// Kernel 4: stable centered ranks of 80 losses -> coef[p] = (r+ - r-)/(79*POP)
// rank[i] = #{j : l[j] < l[i]} + #{j < i : l[j] == l[i]}  (matches jnp.argsort)
// ---------------------------------------------------------------------------
__global__ void __launch_bounds__(128)
k_ranks(const float* __restrict__ losses, float* __restrict__ coef)
{
    __shared__ float l[2 * POP];
    __shared__ float rk[2 * POP];
    const int t = threadIdx.x;
    if (t < 2 * POP) l[t] = losses[t];
    __syncthreads();
    if (t < 2 * POP) {
        float li = l[t];
        int cnt = 0;
        for (int j = 0; j < 2 * POP; ++j) {
            float lj = l[j];
            if (lj < li || (lj == li && j < t)) ++cnt;
        }
        rk[t] = (float)cnt;
    }
    __syncthreads();
    if (t < POP)
        coef[t] = (rk[2 * t] - rk[2 * t + 1]) * (1.0f / (79.0f * (float)POP));
}

// ---------------------------------------------------------------------------
// Kernel 5: grad[2i..2i+1] = sum_p noise[p][2i..2i+1] * coef[p]
// Pure streaming of 339 MB; b64 loads are always 8B-aligned (NP even),
// NP/2 pairs exactly -> no scalar tail.
// ---------------------------------------------------------------------------
__global__ void __launch_bounds__(256)
k_grad(const float* __restrict__ noise, const float* __restrict__ coef,
       float* __restrict__ grad)
{
    __shared__ float c[POP];
    if (threadIdx.x < POP) c[threadIdx.x] = coef[threadIdx.x];
    __syncthreads();
    const int i2 = blockIdx.x * 256 + threadIdx.x;
    if (i2 >= NP / 2) return;
    float gx = 0.0f, gy = 0.0f;
    #pragma unroll 4
    for (int p = 0; p < POP; ++p) {
        if (p + 8 < POP)
            __builtin_prefetch(&noise[(size_t)(p + 8) * NP + 2 * i2], 0, 0);
        v2f nv = *(const v2f*)&noise[(size_t)p * NP + 2 * i2];
        gx += nv.x * c[p];
        gy += nv.y * c[p];
    }
    v2f g = { gx, gy };
    *(v2f*)&grad[2 * i2] = g;
}

// ---------------------------------------------------------------------------
extern "C" void kernel_launch(void* const* d_in, const int* in_sizes, int n_in,
                              void* d_out, int out_size, void* d_ws, size_t ws_size,
                              hipStream_t stream)
{
    const float* x     = (const float*)d_in[0];
    const float* y     = (const float*)d_in[1];
    const float* W1    = (const float*)d_in[2];
    const float* b1    = (const float*)d_in[3];
    const float* W2    = (const float*)d_in[4];
    const float* b2    = (const float*)d_in[5];
    const float* noise = (const float*)d_in[6];
    float* grad = (float*)d_out;

    char* ws = (char*)d_ws;
    float* Hb       = (float*)ws;                                  // 256*2048*4   = 2,097,152 B
    float* predpart = (float*)(ws + (size_t)NROW * HID * 4);       // 80*16*256*10*4 = 13,107,200 B
    float* losses   = predpart + (size_t)2 * POP * 16 * NROW * D_OUT;
    float* coef     = losses + 2 * POP;

    k_base_hidden<<<dim3(HID / 128, NROW / 128), dim3(256), 0, stream>>>(x, W1, Hb);
    k_noise_fused<<<dim3(HID / 128, NROW / 128, POP), dim3(256), 0, stream>>>(
        x, Hb, b1, W2, noise, predpart);
    k_losses<<<dim3(2 * POP), dim3(256), 0, stream>>>(predpart, y, b2, noise, losses);
    k_ranks<<<dim3(1), dim3(128), 0, stream>>>(losses, coef);
    k_grad<<<dim3((NP / 2 + 255) / 256), dim3(256), 0, stream>>>(noise, coef, grad);
}
